// Head_27307402068761
// MI455X (gfx1250) — matrix operand order repaired and checked by reading, not compiled, not run on hardware
//
#include <hip/hip_runtime.h>

// ---------------------------------------------------------------------------
// Flash-attention for B=8, T=4096, C=1024, H=64 on gfx1250 (wave32, WMMA).
// bf16 WMMA (v_wmma_f32_16x16x32_bf16) with fp32 accumulation, fp32 online
// softmax, fp32 output.  x is streamed from HBM exactly once (fused QKV).
// ---------------------------------------------------------------------------

typedef __attribute__((ext_vector_type(16))) __bf16       bfrag; // 32B A/B frag
typedef __attribute__((ext_vector_type(8)))  float        v8f;   // C/D frag
typedef __attribute__((ext_vector_type(4)))  float        v4f;
typedef __attribute__((ext_vector_type(4)))  unsigned int v4u;

union Frag {
    bfrag        v;
    v4u          q[2];
    unsigned int u[8];
};

__device__ __forceinline__ unsigned short f2bf(float f) {
    unsigned int u = __float_as_uint(f);
    return (unsigned short)((u + 0x7FFFu + ((u >> 16) & 1u)) >> 16);  // RNE
}

__device__ __forceinline__ unsigned int pack2bf(float lo, float hi) {
#if __has_builtin(__builtin_amdgcn_cvt_pk_bf16_f32)
    typedef __attribute__((ext_vector_type(2))) __bf16 v2bf;
    union { v2bf v; unsigned int u; } c;
    c.v = __builtin_amdgcn_cvt_pk_bf16_f32(lo, hi);   // v_cvt_pk_bf16_f32
    return c.u;
#else
    return (unsigned int)f2bf(lo) | ((unsigned int)f2bf(hi) << 16);
#endif
}

#define ATT_T 4096
#define ATT_C 1024
#define ATT_H 64

// ---------------------------------------------------------------------------
// Kernel 1: W [1024][64] fp32 -> Wt [3][64][1024] bf16 (transposed so GEMM
// B-fragments are contiguous b128 loads).
// ---------------------------------------------------------------------------
__global__ void attn_wconv(const float* __restrict__ Wq,
                           const float* __restrict__ Wk,
                           const float* __restrict__ Wv,
                           unsigned short* __restrict__ Wt) {
    int tid = blockIdx.x * 256 + threadIdx.x;      // 3*64*1024 = 196608 threads
    int mat = tid >> 16;
    int rem = tid & 0xFFFF;
    int n   = rem >> 10;                           // head col [0,64)
    int k   = rem & 1023;                          // embed row [0,1024)
    const float* W = (mat == 0) ? Wq : ((mat == 1) ? Wk : Wv);
    Wt[tid] = f2bf(W[k * ATT_H + n]);
}

// ---------------------------------------------------------------------------
// Kernel 2: fused QKV projection.  One wave = 16 rows x 64 cols of ALL THREE
// projections (x read once).  12 accumulators; A fragment shared across Q/K/V.
// B fragments of each matrix loaded into distinct registers BEFORE the WMMAs
// so loads overlap compute (staggered s_wait_loadcnt instead of full drains).
// Q,K stored row-major bf16 [B*T][64]; V stored transposed bf16 [B][64][T].
// ---------------------------------------------------------------------------
__global__ __launch_bounds__(32) void attn_qkv_proj(
        const float* __restrict__ x, const unsigned short* __restrict__ Wt,
        unsigned short* __restrict__ qb, unsigned short* __restrict__ kb,
        unsigned short* __restrict__ vT) {
    const int lane = threadIdx.x;
    const int hi   = lane >> 4;       // lane half selects K sub-chunk
    const int ln   = lane & 15;
    const int t0   = blockIdx.x * 16; // 16-row tile of the 32768-row GEMM

    v8f acc[3][4] = {};               // [mat][n-tile]

    for (int k0 = 0; k0 < ATT_C; k0 += 32) {
        // A fragment (16x32 bf16): row = ln; lane half picks K {0..7,16..23}
        // vs {8..15,24..31}.  Load fp32 x, pack to bf16.
        Frag a;
        const float* xrow = x + (size_t)(t0 + ln) * ATT_C + k0 + hi * 8;
        v4f f0 = *(const v4f*)(xrow);
        v4f f1 = *(const v4f*)(xrow + 4);
        v4f f2 = *(const v4f*)(xrow + 16);
        v4f f3 = *(const v4f*)(xrow + 20);
        a.u[0] = pack2bf(f0.x, f0.y);  a.u[1] = pack2bf(f0.z, f0.w);
        a.u[2] = pack2bf(f1.x, f1.y);  a.u[3] = pack2bf(f1.z, f1.w);
        a.u[4] = pack2bf(f2.x, f2.y);  a.u[5] = pack2bf(f2.z, f2.w);
        a.u[6] = pack2bf(f3.x, f3.y);  a.u[7] = pack2bf(f3.z, f3.w);

#pragma unroll
        for (int mat = 0; mat < 3; ++mat) {
            // Load all 4 B fragments first (distinct registers), then compute.
            Frag b[4];
#pragma unroll
            for (int nt = 0; nt < 4; ++nt) {
                const unsigned short* wp =
                    Wt + ((size_t)mat << 16) + (size_t)(nt * 16 + ln) * ATT_C
                       + k0 + hi * 8;
                b[nt].q[0] = *(const v4u*)(wp);
                b[nt].q[1] = *(const v4u*)(wp + 16);
            }
#pragma unroll
            for (int nt = 0; nt < 4; ++nt) {
                acc[mat][nt] = __builtin_amdgcn_wmma_f32_16x16x32_bf16(
                    false, a.v, false, b[nt].v, (short)0, acc[mat][nt],
                    false, false);
            }
        }
    }

    // D layout: VGPR r holds row (hi ? r+8 : r), col = ln.
#pragma unroll
    for (int nt = 0; nt < 4; ++nt) {
#pragma unroll
        for (int r = 0; r < 8; ++r) {
            size_t t = (size_t)(t0 + r + hi * 8);
            int    h = nt * 16 + ln;
            qb[t * ATT_H + h] = f2bf(acc[0][nt][r]);
            kb[t * ATT_H + h] = f2bf(acc[1][nt][r]);
        }
    }
    {   // V transposed: one lane's 8 rows are consecutive tokens -> b128 store
        int bi  = t0 >> 12;
        int tl0 = t0 & (ATT_T - 1);
#pragma unroll
        for (int nt = 0; nt < 4; ++nt) {
            int h = nt * 16 + ln;
            v4u pk;
            pk.x = pack2bf(acc[2][nt][0], acc[2][nt][1]);
            pk.y = pack2bf(acc[2][nt][2], acc[2][nt][3]);
            pk.z = pack2bf(acc[2][nt][4], acc[2][nt][5]);
            pk.w = pack2bf(acc[2][nt][6], acc[2][nt][7]);
            *(v4u*)(vT + ((size_t)bi * ATT_H + h) * ATT_T + tl0 + hi * 8) = pk;
        }
    }
}

// ---------------------------------------------------------------------------
// Kernel 3: causal flash attention.  One wave per 32-query tile (two 16-row
// M-subtiles sharing every K/V B-fragment -> 16 WMMAs per 32-key iteration).
// Online softmax in registers; P staged through a 2KB LDS tile to convert
// D-layout -> A-layout.  Next key tile prefetched (global_prefetch_b8).
// ---------------------------------------------------------------------------
__global__ __launch_bounds__(32) void attn_fwd(
        const unsigned short* __restrict__ qb,
        const unsigned short* __restrict__ kb,
        const unsigned short* __restrict__ vT,
        float* __restrict__ out) {
    __shared__ __align__(16) unsigned short pl[32 * 32];  // P tile bf16

    const int lane = threadIdx.x;
    const int hi   = lane >> 4;
    const int ln   = lane & 15;
    const int b    = blockIdx.y;                 // batch
    const int i0   = blockIdx.x * 32;            // query row base within batch
    const size_t qbase = (size_t)b * ATT_T + i0;

    // Q A-fragments: [m-subtile][head-dim half], loaded once.
    Frag aq[2][2];
#pragma unroll
    for (int mt = 0; mt < 2; ++mt) {
#pragma unroll
        for (int f = 0; f < 2; ++f) {
            const unsigned short* qp =
                qb + (qbase + mt * 16 + ln) * ATT_H + f * 32 + hi * 16;
            aq[mt][f].q[0] = *(const v4u*)(qp);
            aq[mt][f].q[1] = *(const v4u*)(qp + 8);
        }
    }

    v8f o[2][4] = {};                 // 32x64 fp32 output accumulator
    float m[2][8], l[2][8];
#pragma unroll
    for (int mt = 0; mt < 2; ++mt)
#pragma unroll
        for (int r = 0; r < 8; ++r) { m[mt][r] = -INFINITY; l[mt][r] = 0.0f; }

    const float scale = 0.125f * 1.44269504088896340736f; // 1/sqrt(64)*log2(e)
    const int qend = i0 + 31;

    for (int j0 = 0; j0 <= qend; j0 += 32) {
        // ---- prefetch next key tile while this one computes ----
        if (j0 + 32 <= qend) {
            __builtin_prefetch(
                kb + ((size_t)b * ATT_T + j0 + 32 + lane) * ATT_H, 0, 3);
#pragma unroll
            for (int nt = 0; nt < 4; ++nt)
                __builtin_prefetch(
                    vT + ((size_t)b * ATT_H + nt * 16 + ln) * ATT_T
                       + j0 + 32 + hi * 16, 0, 3);
        }

        // ---- K B-fragments, shared by both M-subtiles ----
        Frag bk[2][2];   // [j-subtile][head-dim half]
#pragma unroll
        for (int jt = 0; jt < 2; ++jt) {
            const unsigned short* kp =
                kb + ((size_t)b * ATT_T + j0 + jt * 16 + ln) * ATT_H + hi * 16;
            bk[jt][0].q[0] = *(const v4u*)(kp);
            bk[jt][0].q[1] = *(const v4u*)(kp + 8);
            bk[jt][1].q[0] = *(const v4u*)(kp + 32);
            bk[jt][1].q[1] = *(const v4u*)(kp + 40);
        }

        // ---- S = Q K^T : 8 WMMAs ----
        v8f s[2][2];
#pragma unroll
        for (int mt = 0; mt < 2; ++mt) {
#pragma unroll
            for (int jt = 0; jt < 2; ++jt) {
                v8f z = {};
                v8f t = __builtin_amdgcn_wmma_f32_16x16x32_bf16(
                    false, aq[mt][0].v, false, bk[jt][0].v, (short)0, z,
                    false, false);
                s[mt][jt] = __builtin_amdgcn_wmma_f32_16x16x32_bf16(
                    false, aq[mt][1].v, false, bk[jt][1].v, (short)0, t,
                    false, false);
            }
        }

        // ---- causal mask + online softmax (rows live in 16-lane halves) ----
#pragma unroll
        for (int mt = 0; mt < 2; ++mt) {
#pragma unroll
            for (int r = 0; r < 8; ++r) {
                int grow = i0 + mt * 16 + r + hi * 8;
                float s0 = s[mt][0][r] * scale;
                float s1 = s[mt][1][r] * scale;
                if (j0 + ln > grow)       s0 = -3.0e38f;
                if (j0 + 16 + ln > grow)  s1 = -3.0e38f;
                float mx = fmaxf(s0, s1);
#pragma unroll
                for (int off = 1; off < 16; off <<= 1)
                    mx = fmaxf(mx, __shfl_xor(mx, off, 32));
                float mn    = fmaxf(m[mt][r], mx);
                float alpha = exp2f(m[mt][r] - mn);  // exp2(-inf)=0 first tile
                float p0    = exp2f(s0 - mn);
                float p1    = exp2f(s1 - mn);
                float rs    = p0 + p1;
#pragma unroll
                for (int off = 1; off < 16; off <<= 1)
                    rs += __shfl_xor(rs, off, 32);
                l[mt][r] = l[mt][r] * alpha + rs;
                m[mt][r] = mn;
#pragma unroll
                for (int nt = 0; nt < 4; ++nt) o[mt][nt][r] *= alpha;
                int lrow = mt * 16 + r + hi * 8;     // D-layout row
                pl[lrow * 32 + ln]      = f2bf(p0);
                pl[lrow * 32 + 16 + ln] = f2bf(p1);
            }
        }
        __syncthreads();   // single-wave block: orders LDS writes vs reads

        // ---- P A-fragments from LDS, V B-fragments from global ----
        Frag ap[2];
#pragma unroll
        for (int mt = 0; mt < 2; ++mt) {
            const unsigned short* pp = pl + (mt * 16 + ln) * 32 + hi * 8;
            ap[mt].q[0] = *(const v4u*)(pp);
            ap[mt].q[1] = *(const v4u*)(pp + 16);
        }
        Frag bv[4];      // distinct registers: loads overlap the WMMAs below
#pragma unroll
        for (int nt = 0; nt < 4; ++nt) {
            const unsigned short* vp =
                vT + ((size_t)b * ATT_H + nt * 16 + ln) * ATT_T + j0 + hi * 8;
            bv[nt].q[0] = *(const v4u*)(vp);
            bv[nt].q[1] = *(const v4u*)(vp + 16);
        }

        // ---- O += P V : 8 WMMAs ----
#pragma unroll
        for (int nt = 0; nt < 4; ++nt) {
#pragma unroll
            for (int mt = 0; mt < 2; ++mt) {
                o[mt][nt] = __builtin_amdgcn_wmma_f32_16x16x32_bf16(
                    false, ap[mt].v, false, bv[nt].v, (short)0, o[mt][nt],
                    false, false);
            }
        }
        __syncthreads();   // protect pl before next iteration rewrites it
    }

    // ---- normalize by row sum, store fp32 ----
#pragma unroll
    for (int mt = 0; mt < 2; ++mt) {
#pragma unroll
        for (int nt = 0; nt < 4; ++nt) {
#pragma unroll
            for (int r = 0; r < 8; ++r) {
                size_t row = qbase + mt * 16 + r + hi * 8;
                out[row * ATT_H + nt * 16 + ln] = o[mt][nt][r] / l[mt][r];
            }
        }
    }
}

// ---------------------------------------------------------------------------
// Launch
// ---------------------------------------------------------------------------
extern "C" void kernel_launch(void* const* d_in, const int* in_sizes, int n_in,
                              void* d_out, int out_size, void* d_ws, size_t ws_size,
                              hipStream_t stream) {
    const float* x  = (const float*)d_in[0];   // [8,4096,1024]
    const float* Wq = (const float*)d_in[1];   // [1024,64]
    const float* Wk = (const float*)d_in[2];
    const float* Wv = (const float*)d_in[3];
    float* out = (float*)d_out;                // [8,4096,64]

    const size_t BT = (size_t)8 * ATT_T;       // 32768 tokens
    unsigned short* qbuf = (unsigned short*)d_ws;          // [32768][64] bf16
    unsigned short* kbuf = qbuf + BT * ATT_H;              // [32768][64] bf16
    unsigned short* vT   = kbuf + BT * ATT_H;              // [8][64][4096] bf16
    unsigned short* Wt   = vT   + BT * ATT_H;              // [3][64][1024] bf16
    // total workspace: (3*2097152 + 196608) * 2 bytes ~= 13.0 MB

    attn_wconv<<<768, 256, 0, stream>>>(Wq, Wk, Wv, Wt);
    attn_qkv_proj<<<2048, 32, 0, stream>>>(x, Wt, qbuf, kbuf, vT);
    attn_fwd<<<dim3(128, 8), 32, 0, stream>>>(qbuf, kbuf, vT, out);
}
